// AttentionLSTM_85521388798688
// MI455X (gfx1250) — compile-verified, matrix operand order checked by reading, not run on hardware
//
#include <hip/hip_runtime.h>
#include <hip/hip_bf16.h>
#include <math.h>

// Problem sizes (fixed by the reference)
#define NB   256          // batch
#define TT   64           // timesteps
#define DD   1024         // input dim
#define HH   1024         // hidden dim
#define KTOT 3072         // fused K = H (h) + H (attn) + D (x)
#define NOUT 4096         // 4*H gate columns

typedef __attribute__((ext_vector_type(16))) __bf16 v16bf;
typedef __attribute__((ext_vector_type(8)))  __bf16 v8bf;
typedef __attribute__((ext_vector_type(8)))  float  v8f;

union Frag16 { v16bf v; v8bf h[2]; };

// ---------------------------------------------------------------------------
// One-time: pack [Wh; Wattn; Wx] transposed to bf16: wt[n][k], n<4096, k<3072.
// Column-major storage makes B-fragment LDS loads contiguous in the GEMM.
// ---------------------------------------------------------------------------
__global__ __launch_bounds__(256)
void pack_weights(const float* __restrict__ Wx, const float* __restrict__ Wh,
                  const float* __restrict__ Wattn, __bf16* __restrict__ wt) {
    size_t idx = (size_t)blockIdx.x * 256 + threadIdx.x;   // 4096*3072 total
    int nn = (int)(idx / KTOT);
    int k  = (int)(idx % KTOT);
    float v;
    if (k < HH)            v = Wh[(size_t)k * NOUT + nn];
    else if (k < 2 * HH)   v = Wattn[(size_t)(k - HH) * NOUT + nn];
    else                   v = Wx[(size_t)(k - 2 * HH) * NOUT + nn];
    wt[idx] = (__bf16)v;
}

// ---------------------------------------------------------------------------
// One-time: h0 = c0 = mean over the 16 spatial locations of A.
// ---------------------------------------------------------------------------
__global__ __launch_bounds__(256)
void init_hc(const float* __restrict__ Aflat, float* __restrict__ h,
             float* __restrict__ c) {
    int idx = blockIdx.x * 256 + threadIdx.x;              // n*1024 + j
    const float* a = Aflat + (size_t)idx * 16;
    float s = 0.f;
#pragma unroll
    for (int l = 0; l < 16; ++l) s += a[l];
    s *= (1.0f / 16.0f);
    h[idx] = s;
    c[idx] = s;
}

// ---------------------------------------------------------------------------
// Per step: attention (softmax over 16 locations) + pack [h|attn|xt] -> bf16.
// One block per batch row n; 256 threads, each owns 4 hidden units.
// ---------------------------------------------------------------------------
__global__ __launch_bounds__(256)
void attn_pack(const float* __restrict__ x,      // (N,T,D)
               const float* __restrict__ Aflat,  // (N,H,16)
               const float* __restrict__ h,      // (N,H)
               __bf16* __restrict__ act,         // (N,3072)
               int t) {
    const int n = blockIdx.x;
    const int tid = threadIdx.x;
    __shared__ float sdot[16][256];
    __shared__ float w[16];

    const float scale = 1.0f / 32.0f;  // 1/sqrt(1024)

    float hv[4];
    float partial[16];
#pragma unroll
    for (int l = 0; l < 16; ++l) partial[l] = 0.f;

#pragma unroll
    for (int j = 0; j < 4; ++j) {
        int hh = tid + j * 256;
        float hvj = h[(size_t)n * HH + hh];
        hv[j] = hvj;
        const float* arow = Aflat + ((size_t)n * HH + hh) * 16;
#pragma unroll
        for (int l = 0; l < 16; ++l) partial[l] += hvj * arow[l];
    }
#pragma unroll
    for (int l = 0; l < 16; ++l) sdot[l][tid] = partial[l];
    __syncthreads();

    if (tid < 16) {                 // reduce 256 partials for logit `tid`
        float s = 0.f;
        for (int k = 0; k < 256; ++k) s += sdot[tid][k];
        sdot[tid][0] = s * scale;
    }
    __syncthreads();

    if (tid == 0) {                 // 16-way softmax, serial (cheap)
        float m = -1e30f;
#pragma unroll
        for (int l = 0; l < 16; ++l) m = fmaxf(m, sdot[l][0]);
        float e[16], sum = 0.f;
#pragma unroll
        for (int l = 0; l < 16; ++l) { e[l] = __expf(sdot[l][0] - m); sum += e[l]; }
        float inv = 1.0f / sum;
#pragma unroll
        for (int l = 0; l < 16; ++l) w[l] = e[l] * inv;
    }
    __syncthreads();

#pragma unroll
    for (int j = 0; j < 4; ++j) {
        int hh = tid + j * 256;
        const float* arow = Aflat + ((size_t)n * HH + hh) * 16;
        float at = 0.f;
#pragma unroll
        for (int l = 0; l < 16; ++l) at += arow[l] * w[l];
        __bf16* dst = act + (size_t)n * KTOT;
        dst[hh]          = (__bf16)hv[j];
        dst[HH + hh]     = (__bf16)at;
        dst[2 * HH + hh] = (__bf16)x[((size_t)n * TT + t) * DD + hh];
    }
}

// ---------------------------------------------------------------------------
// Per step fused GEMM (split-K): gpart[z] = act(256x3072) x wt^T over K-slice z.
// BM=BN=128 tile, 8 waves (2x4), each wave -> 4x2 WMMA 16x16 tiles.
// KSPLIT=4 slices of K=768 -> 2 x 32 x 4 = 256 workgroups per step.
// Double-buffered LDS with register staging: one barrier per K-chunk; the
// next chunk's global_load_b128s are in flight while 8 WMMAs execute.
// ---------------------------------------------------------------------------
#define BM 128
#define BN 128
#define KC 32
#define KSPLIT 4
#define KPER (KTOT / KSPLIT)   // 768

__global__ __launch_bounds__(256)
void gemm_step(const __bf16* __restrict__ act,  // (256,3072) row-major
               const __bf16* __restrict__ wt,   // (4096,3072) row-major (= W^T)
               float* __restrict__ gpart) {     // (KSPLIT,256,4096)
    __shared__ __align__(16) __bf16 lA[2][BM][KC];  // 16 KB
    __shared__ __align__(16) __bf16 lB[2][BN][KC];  // 16 KB

    const int tid  = threadIdx.x;
    const int lane = tid & 31;
    const int wave = tid >> 5;          // 0..7
    const int wm   = wave >> 2;         // 0..1  : 64-row M slab
    const int wn   = wave & 3;          // 0..3  : 32-col N slab
    const int blockM = blockIdx.x * BM; // gridDim.x = 2
    const int blockN = blockIdx.y * BN; // gridDim.y = 32
    const int kbase  = blockIdx.z * KPER;

    const int l15   = lane & 15;
    const int khalf = (lane >> 4) * 8;  // 0 or 8: per-lane K sub-range

    v8f acc[4][2];
#pragma unroll
    for (int mt = 0; mt < 4; ++mt)
#pragma unroll
        for (int nt = 0; nt < 2; ++nt)
            acc[mt][nt] = (v8f)(0.0f);

    // cooperative-load coordinates: 2 threads per row, 16 bf16 each
    const int ldr = tid >> 1;           // 0..127
    const int ldc = (tid & 1) * 16;     // 0 or 16
    const __bf16* gA = act + (size_t)(blockM + ldr) * KTOT + kbase + ldc;
    const __bf16* gB = wt  + (size_t)(blockN + ldr) * KTOT + kbase + ldc;

    // preload chunk 0 into LDS buffer 0
    uint4 ra0 = ((const uint4*)gA)[0];
    uint4 ra1 = ((const uint4*)gA)[1];
    uint4 rb0 = ((const uint4*)gB)[0];
    uint4 rb1 = ((const uint4*)gB)[1];
    *(uint4*)&lA[0][ldr][ldc]     = ra0;
    *(uint4*)&lA[0][ldr][ldc + 8] = ra1;
    *(uint4*)&lB[0][ldr][ldc]     = rb0;
    *(uint4*)&lB[0][ldr][ldc + 8] = rb1;
    __syncthreads();

    for (int kk = 0; kk < KPER; kk += KC) {
        const int cur = (kk / KC) & 1;
        const bool more = (kk + KC) < KPER;

        // issue next chunk's global loads now; they retire under the WMMAs
        if (more) {
            const __bf16* pa = gA + kk + KC;
            const __bf16* pb = gB + kk + KC;
            ra0 = ((const uint4*)pa)[0];
            ra1 = ((const uint4*)pa)[1];
            rb0 = ((const uint4*)pb)[0];
            rb1 = ((const uint4*)pb)[1];
            __builtin_prefetch(pa + KC, 0, 1);   // global_prefetch_b8, 2 ahead
            __builtin_prefetch(pb + KC, 0, 1);
        }

        // per-lane fragments: two contiguous 16B LDS reads each (K split
        // low/high per lane half, matching the bf16 WMMA operand layout)
        Frag16 afr[4], bfr[2];
#pragma unroll
        for (int mt = 0; mt < 4; ++mt) {
            const __bf16* ap = &lA[cur][wm * 64 + mt * 16 + l15][0];
            afr[mt].h[0] = *(const v8bf*)(ap + khalf);
            afr[mt].h[1] = *(const v8bf*)(ap + 16 + khalf);
        }
#pragma unroll
        for (int nt = 0; nt < 2; ++nt) {
            const __bf16* bp = &lB[cur][wn * 32 + nt * 16 + l15][0];
            bfr[nt].h[0] = *(const v8bf*)(bp + khalf);
            bfr[nt].h[1] = *(const v8bf*)(bp + 16 + khalf);
        }

#pragma unroll
        for (int mt = 0; mt < 4; ++mt)
#pragma unroll
            for (int nt = 0; nt < 2; ++nt)
                acc[mt][nt] = __builtin_amdgcn_wmma_f32_16x16x32_bf16(
                    /*neg_a=*/false, afr[mt].v,
                    /*neg_b=*/false, bfr[nt].v,
                    /*c_mod=*/(short)0, acc[mt][nt],
                    /*reuse_a=*/false, /*reuse_b=*/false);

        // stage next chunk into the other buffer; reads above hit `cur`,
        // writes hit `cur^1`, so one barrier per iteration is race-free
        if (more) {
            *(uint4*)&lA[cur ^ 1][ldr][ldc]     = ra0;
            *(uint4*)&lA[cur ^ 1][ldr][ldc + 8] = ra1;
            *(uint4*)&lB[cur ^ 1][ldr][ldc]     = rb0;
            *(uint4*)&lB[cur ^ 1][ldr][ldc + 8] = rb1;
            __syncthreads();
        }
    }

    // epilogue: C/D layout — VGPR r holds row (lane>=16 ? 8+r : r), col = lane&15
    float* g = gpart + (size_t)blockIdx.z * NB * NOUT;
    const int rbase = blockM + wm * 64 + (lane >> 4) * 8;
    const int cbase = blockN + wn * 32 + l15;
#pragma unroll
    for (int mt = 0; mt < 4; ++mt)
#pragma unroll
        for (int nt = 0; nt < 2; ++nt)
#pragma unroll
            for (int r = 0; r < 8; ++r)
                g[(size_t)(rbase + mt * 16 + r) * NOUT + cbase + nt * 16] =
                    acc[mt][nt][r];
}

// ---------------------------------------------------------------------------
// Per step: reduce split-K partials + bias + LSTM cell update; writes out[:,t,:].
// Deterministic reduction (fixed order) -> bit-stable across graph replays.
// ---------------------------------------------------------------------------
__global__ __launch_bounds__(256)
void lstm_update(const float* __restrict__ gpart, const float* __restrict__ b,
                 float* __restrict__ h, float* __restrict__ c,
                 float* __restrict__ out, int t) {
    int idx = blockIdx.x * 256 + threadIdx.x;  // n*1024 + j
    int n = idx >> 10;
    int j = idx & 1023;

    float a4[4];
#pragma unroll
    for (int gix = 0; gix < 4; ++gix) {
        int col = j + gix * HH;
        float s = b[col];
#pragma unroll
        for (int z = 0; z < KSPLIT; ++z)
            s += gpart[((size_t)z * NB + n) * NOUT + col];
        a4[gix] = s;
    }

    float gi = 1.0f / (1.0f + __expf(-a4[0]));
    float gf = 1.0f / (1.0f + __expf(-a4[1]));
    float go = 1.0f / (1.0f + __expf(-a4[2]));
    float gg = tanhf(a4[3]);
    float cn = gf * c[idx] + gi * gg;
    c[idx] = cn;
    float hn = go * tanhf(cn);
    h[idx] = hn;
    out[((size_t)n * TT + t) * HH + j] = hn;
}

// ---------------------------------------------------------------------------
extern "C" void kernel_launch(void* const* d_in, const int* in_sizes, int n_in,
                              void* d_out, int out_size, void* d_ws, size_t ws_size,
                              hipStream_t stream) {
    (void)in_sizes; (void)n_in; (void)out_size; (void)ws_size;

    const float* x     = (const float*)d_in[0];   // (N,T,D)
    const float* A     = (const float*)d_in[1];   // (N,H,4,4) == (N,H,16)
    const float* Wx    = (const float*)d_in[2];   // (D,4H)
    const float* Wh    = (const float*)d_in[3];   // (H,4H)
    const float* Wattn = (const float*)d_in[4];   // (H,4H)
    const float* b     = (const float*)d_in[5];   // (4H)
    float* out         = (float*)d_out;           // (N,T,H)

    // workspace carve-out (~43.5 MB total)
    char* ws = (char*)d_ws;
    __bf16* wt  = (__bf16*)ws;  ws += (size_t)NOUT * KTOT * sizeof(__bf16);          // 24 MB
    __bf16* act = (__bf16*)ws;  ws += (size_t)NB * KTOT * sizeof(__bf16);            // 1.5 MB
    float* h    = (float*)ws;   ws += (size_t)NB * HH * sizeof(float);               // 1 MB
    float* c    = (float*)ws;   ws += (size_t)NB * HH * sizeof(float);               // 1 MB
    float* gp   = (float*)ws;   ws += (size_t)KSPLIT * NB * NOUT * sizeof(float);    // 16 MB

    // one-time setup
    pack_weights<<<(int)(((size_t)NOUT * KTOT) / 256), 256, 0, stream>>>(Wx, Wh, Wattn, wt);
    init_hc<<<(NB * HH) / 256, 256, 0, stream>>>(A, h, c);

    // sequential recurrence: 3 kernels per timestep, stream-ordered
    for (int t = 0; t < TT; ++t) {
        attn_pack<<<NB, 256, 0, stream>>>(x, A, h, act, t);
        gemm_step<<<dim3(NB / BM, NOUT / BN, KSPLIT), 256, 0, stream>>>(act, wt, gp);
        lstm_update<<<(NB * HH) / 256, 256, 0, stream>>>(gp, b, h, c, out, t);
    }
}
